// RWKV6_CrossAttention_78391743086639
// MI455X (gfx1250) — matrix operand order repaired and checked by reading, not compiled, not run on hardware
//
#include <hip/hip_runtime.h>
#include <hip/hip_bf16.h>
#include <math.h>

// ---------------------------------------------------------------------------
// RWKV6 cross-attention for MI455X (gfx1250, wave32, WMMA bf16 16x16x32).
//   - weights pre-converted to bf16 once (no per-tile cvt work)
//   - async global->LDS loads (ASYNCcnt) for the sequential scan staging
//   - native v_tanh_f32 (branch-free) for activation epilogues
// ---------------------------------------------------------------------------

#define HID 256
#define NH  4
#define DK  64
#define DV  64
#define KD  256
#define VD  256
#define RLR 32
#define GLR 64
#define B_  4
#define LQ  32
#define LK  128
#define EB  (B_*LQ)
#define EPSV 1e-5f

typedef __bf16 bf16_t;
typedef bf16_t v16bf __attribute__((ext_vector_type(16)));
typedef float  v8f   __attribute__((ext_vector_type(8)));

// ---- bf16 weight workspace layout (elements, after kbuf/vbuf f32 region) ----
#define WSZ_KV0  (GLR*HID)        // 16384
#define WSZ_KV2  (HID*2*RLR)      // 16384
#define WSZ_K    (KD*HID)         // 65536
#define WSZ_V    (VD*HID)         // 65536
#define WSZ_RWG0 (3*RLR*HID)      // 24576
#define WSZ_RWG2 (HID*3*RLR)      // 24576
#define WSZ_W1   (GLR*HID)        // 16384
#define WSZ_W2   (KD*GLR)         // 16384
#define OFF_KV0  0
#define OFF_KV2  (OFF_KV0 + WSZ_KV0)
#define OFF_K    (OFF_KV2 + WSZ_KV2)
#define OFF_V    (OFF_K   + WSZ_K)
#define OFF_RWG0 (OFF_V   + WSZ_V)
#define OFF_RWG2 (OFF_RWG0 + WSZ_RWG0)
#define OFF_W1   (OFF_RWG2 + WSZ_RWG2)
#define OFF_W2   (OFF_W1  + WSZ_W1)
#define WBF_TOTAL (OFF_W2 + WSZ_W2)   // 245760 elements

// branch-free tanh: native V_TANH_F32 if the builtin exists, else exp-based.
__device__ __forceinline__ float fast_tanh(float x) {
#if __has_builtin(__builtin_amdgcn_tanhf)
  return __builtin_amdgcn_tanhf(x);
#else
  float e = __expf(2.f * x);
  return 1.f - 2.f / (e + 1.f);
#endif
}

// A fragment: 16x32 bf16 tile, row-major in LDS (stride lda elements).
// k order used here (contiguous, same as ldb_frag): lane&15 = M row; lanes 0-15 hold K 0-15, lanes 16-31 hold K 16-31.
__device__ __forceinline__ v16bf lda_frag(const bf16_t* A, int lda, int mbase,
                                          int kbase, int lane) {
  const int m  = mbase + (lane & 15);
  const int kh = (lane >> 4) << 4;               // 0 or 16
  const bf16_t* row = A + m * lda + kbase;
  v16bf a;
#pragma unroll
  for (int e = 0; e < 16; ++e) a[e] = row[kh + e];
  return a;
}

// B fragment: 32x16 (B[k][n] = W[n][k]); W row-major NxK bf16 in global.
// lane&15 = N col; lanes 0-15 hold K 0-15, lanes 16-31 hold K 16-31.
__device__ __forceinline__ v16bf ldb_frag(const bf16_t* W, int ldk, int nbase,
                                          int kbase, int lane) {
  const int n  = nbase + (lane & 15);
  const int kh = (lane >> 4) << 4;               // 0 or 16
  const bf16_t* row = W + n * ldk + kbase + kh;  // 32B-aligned for all our ldk
  v16bf b;
#pragma unroll
  for (int e = 0; e < 16; ++e) b[e] = row[e];
  return b;
}

__device__ __forceinline__ v8f wmma_bf16(v16bf a, v16bf b, v8f c) {
  // (neg_a, A, neg_b, B, c_mod, C, reuse_a, reuse_b)
  return __builtin_amdgcn_wmma_f32_16x16x32_bf16(false, a, false, b,
                                                 (short)0, c, false, false);
}

// async global -> LDS staging (GLOBAL_LOAD_ASYNC_TO_LDS_B32, GV addressing)
__device__ __forceinline__ void async_g2l_b32(void* lds_dst, const float* gsrc) {
  unsigned lds_off = (unsigned)(uintptr_t)lds_dst;            // LDS aperture low bits
  unsigned long long ga = (unsigned long long)(uintptr_t)gsrc;
  asm volatile("global_load_async_to_lds_b32 %0, %1, off"
               :: "v"(lds_off), "v"(ga) : "memory");
}
__device__ __forceinline__ void wait_asynccnt0() {
  asm volatile("s_wait_asynccnt 0x0" ::: "memory");
}

// ===========================================================================
// Kernel 0: one-time f32 -> bf16 weight conversion into workspace.
// ===========================================================================
__global__ void __launch_bounds__(256) cvt_weights_kernel(
    const float* __restrict__ W_kv0,  const float* __restrict__ W_kv2,
    const float* __restrict__ W_k,    const float* __restrict__ W_v,
    const float* __restrict__ W_rwg0, const float* __restrict__ W_rwg2,
    const float* __restrict__ W_w1,   const float* __restrict__ W_w2,
    bf16_t* __restrict__ wbf)
{
  int idx = blockIdx.x * 256 + threadIdx.x;
  if (idx >= WBF_TOTAL) return;
  float v;
  if      (idx < OFF_KV2)  v = W_kv0 [idx - OFF_KV0];
  else if (idx < OFF_K)    v = W_kv2 [idx - OFF_KV2];
  else if (idx < OFF_V)    v = W_k   [idx - OFF_K];
  else if (idx < OFF_RWG0) v = W_v   [idx - OFF_V];
  else if (idx < OFF_RWG2) v = W_rwg0[idx - OFF_RWG0];
  else if (idx < OFF_W1)   v = W_rwg2[idx - OFF_RWG2];
  else if (idx < OFF_W2)   v = W_w1  [idx - OFF_W1];
  else                     v = W_w2  [idx - OFF_W2];
  wbf[idx] = (bf16_t)v;
}

// ===========================================================================
// Kernel 1: per-batch k/v projections (shared by all 32 lq replicas).
// ===========================================================================
__global__ void __launch_bounds__(256) kv_proj_kernel(
    const float* __restrict__ keyval, const float* __restrict__ mu_kv,
    const float* __restrict__ kv_bias,const bf16_t* __restrict__ wbf,
    float* __restrict__ kbuf,         float* __restrict__ vbuf)
{
  extern __shared__ char smem[];
  bf16_t* Xk  = (bf16_t*)smem;            // 128*256 bf16 (first used as Xs)
  bf16_t* Xv  = Xk + LK*HID;              // 128*256 bf16
  bf16_t* Tkv = Xv + LK*HID;              // 128*64 bf16

  const bf16_t* Wb_kv0 = wbf + OFF_KV0;
  const bf16_t* Wb_kv2 = wbf + OFF_KV2;
  const bf16_t* Wb_k   = wbf + OFF_K;
  const bf16_t* Wb_v   = wbf + OFF_V;

  const int tid  = threadIdx.x;
  const int lane = tid & 31;
  const int wv   = tid >> 5;
  const int bb   = blockIdx.x;
  const float* xb = keyval + (size_t)bb * LK * HID;

  for (int idx = tid; idx < LK*HID; idx += 256) {
    int m = idx >> 8, c = idx & 255;
    float xv = xb[m*HID + c];
    float sh = (m == 0) ? 0.f : xb[(m-1)*HID + c];
    Xk[idx] = (bf16_t)(xv + (sh - xv) * mu_kv[c]);
  }
  __syncthreads();

  const int mstrip = wv * 16;

  // GEMM A: Tkv = tanh(Xs @ W_kv0^T)   (128 x 64)
  for (int nt = 0; nt < 4; ++nt) {
    v8f acc = {};
#pragma unroll
    for (int kb = 0; kb < HID; kb += 32) {
      v16bf a = lda_frag(Xk, HID, mstrip, kb, lane);
      v16bf b = ldb_frag(Wb_kv0, HID, nt*16, kb, lane);
      acc = wmma_bf16(a, b, acc);
    }
    int n = nt*16 + (lane & 15);
#pragma unroll
    for (int j = 0; j < 8; ++j) {
      int m = mstrip + ((lane >> 4) << 3) + j;
      Tkv[m*GLR + n] = (bf16_t)fast_tanh(acc[j]);
    }
  }
  __syncthreads();

  // GEMM B: low-rank expansion -> X̂k / X̂v (bf16 in LDS)
  for (int comp = 0; comp < 2; ++comp) {
    for (int nt = 0; nt < 16; ++nt) {
      v16bf a = lda_frag(Tkv, GLR, mstrip, comp*RLR, lane);
      v16bf b = ldb_frag(Wb_kv2 + comp*RLR, 2*RLR, nt*16, 0, lane);
      v8f acc = {};
      acc = wmma_bf16(a, b, acc);
      int n = nt*16 + (lane & 15);
#pragma unroll
      for (int j = 0; j < 8; ++j) {
        int m = mstrip + ((lane >> 4) << 3) + j;
        float c0 = acc[j] + kv_bias[comp*HID + n];
        float xv = xb[m*HID + n];
        float sh = (m == 0) ? 0.f : xb[(m-1)*HID + n];
        float val = xv + (sh - xv) * c0;
        if (comp == 0) Xk[m*HID + n] = (bf16_t)val;
        else           Xv[m*HID + n] = (bf16_t)val;
      }
    }
  }
  __syncthreads();

  // GEMM C: k = X̂k @ W_k^T -> kbuf
  for (int nt = 0; nt < 16; ++nt) {
    v8f acc = {};
#pragma unroll
    for (int kb = 0; kb < HID; kb += 32) {
      v16bf a = lda_frag(Xk, HID, mstrip, kb, lane);
      v16bf b = ldb_frag(Wb_k, HID, nt*16, kb, lane);
      acc = wmma_bf16(a, b, acc);
    }
    int n = nt*16 + (lane & 15);
#pragma unroll
    for (int j = 0; j < 8; ++j) {
      int m = mstrip + ((lane >> 4) << 3) + j;
      kbuf[((size_t)bb*LK + m)*KD + n] = acc[j];
    }
  }
  // GEMM D: v = X̂v @ W_v^T -> vbuf
  for (int nt = 0; nt < 16; ++nt) {
    v8f acc = {};
#pragma unroll
    for (int kb = 0; kb < HID; kb += 32) {
      v16bf a = lda_frag(Xv, HID, mstrip, kb, lane);
      v16bf b = ldb_frag(Wb_v, HID, nt*16, kb, lane);
      acc = wmma_bf16(a, b, acc);
    }
    int n = nt*16 + (lane & 15);
#pragma unroll
    for (int j = 0; j < 8; ++j) {
      int m = mstrip + ((lane >> 4) << 3) + j;
      vbuf[((size_t)bb*LK + m)*VD + n] = acc[j];
    }
  }
}

// ===========================================================================
// Kernel 2: one workgroup per (b,lq) pair. Decay projection chain (WMMA),
// r/g GEMVs at t=127, register-resident RWKV6 scan, group-norm+silu, W_o GEMV.
// ===========================================================================
__global__ void __launch_bounds__(256) rwkv_fused_kernel(
    const float* __restrict__ query,  const float* __restrict__ keyval,
    const float* __restrict__ mu_rwg, const float* __restrict__ W_rwg2,
    const float* __restrict__ rwg_bias,
    const float* __restrict__ W_r,    const float* __restrict__ b_w2,
    const float* __restrict__ W_g,    const float* __restrict__ u,
    const float* __restrict__ gamma,  const float* __restrict__ beta,
    const float* __restrict__ W_o,    const bf16_t* __restrict__ wbf,
    const float* __restrict__ kbuf,   const float* __restrict__ vbuf,
    float* __restrict__ out)
{
  extern __shared__ char smem[];
  bf16_t* Xh     = (bf16_t*)smem;                        // 128*256 bf16 = 64 KB
  float*  decayl = (float*)(smem + 65536);               // 128*256 f32 = 128 KB
  bf16_t* Tbuf   = (bf16_t*)(smem + 65536 + 131072);     // 128*64 bf16 = 16 KB
  float*  mrow   = (float*)(smem + 65536 + 131072 + 16384);
  float*  t0last = mrow;            // 96 (padded to 128)
  float*  xr     = mrow + 128;      // 256
  float*  xg     = xr + 256;        // 256
  float*  rrow   = xg + 256;        // 256
  float*  sg     = rrow + 256;      // 256
  float*  uu     = sg + 256;        // 256
  float*  ktls   = uu + 256;        // 256
  float*  vtls   = ktls + 256;      // 256
  float*  orow   = vtls + 256;      // 256
  float*  orow2  = orow + 256;      // 256

  const bf16_t* Wb_rwg0 = wbf + OFF_RWG0;
  const bf16_t* Wb_rwg2 = wbf + OFF_RWG2;
  const bf16_t* Wb_w1   = wbf + OFF_W1;
  const bf16_t* Wb_w2   = wbf + OFF_W2;

  const int tid  = threadIdx.x;
  const int lane = tid & 31;
  const int wv   = tid >> 5;
  const int eb   = blockIdx.x;
  const int bb   = eb >> 5;
  const float* xb = keyval + (size_t)bb * LK * HID;
  const float* qr = query  + (size_t)eb * HID;

  // X̂ = x + (q - x) * mu_rwg   (bf16 in LDS)
  for (int idx = tid; idx < LK*HID; idx += 256) {
    int m = idx >> 8, c = idx & 255;
    float xv = xb[m*HID + c];
    Xh[idx] = (bf16_t)(xv + (qr[c] - xv) * mu_rwg[c]);
  }
  __syncthreads();

  const int mstrip = wv * 16;

  // GEMM1: T0 = tanh(X̂ @ W_rwg0^T) (128x96); keep w-slice + full last row
  for (int nt = 0; nt < 6; ++nt) {
    v8f acc = {};
#pragma unroll
    for (int kb = 0; kb < HID; kb += 32) {
      v16bf a = lda_frag(Xh, HID, mstrip, kb, lane);
      v16bf b = ldb_frag(Wb_rwg0, HID, nt*16, kb, lane);
      acc = wmma_bf16(a, b, acc);
    }
    int n = nt*16 + (lane & 15);
#pragma unroll
    for (int j = 0; j < 8; ++j) {
      int m = mstrip + ((lane >> 4) << 3) + j;
      float val = fast_tanh(acc[j]);
      if (n >= RLR && n < 2*RLR) Tbuf[m*RLR + (n - RLR)] = (bf16_t)val;
      if (m == LK-1) t0last[n] = val;
    }
  }
  __syncthreads();

  // GEMM2: w0 = T0w @ W2w^T + b ; X̂ <- x + (q-x)*w0
  for (int nt = 0; nt < 16; ++nt) {
    v16bf a = lda_frag(Tbuf, RLR, mstrip, 0, lane);
    v16bf b = ldb_frag(Wb_rwg2 + RLR, 3*RLR, nt*16, 0, lane);
    v8f acc = {};
    acc = wmma_bf16(a, b, acc);
    int n = nt*16 + (lane & 15);
#pragma unroll
    for (int j = 0; j < 8; ++j) {
      int m = mstrip + ((lane >> 4) << 3) + j;
      float w0 = acc[j] + rwg_bias[HID + n];
      float xv = xb[m*HID + n];
      Xh[m*HID + n] = (bf16_t)(xv + (qr[n] - xv) * w0);
    }
  }
  __syncthreads();

  // GEMM3: T1 = tanh(X̂ @ W_w1^T) (128x64)
  for (int nt = 0; nt < 4; ++nt) {
    v8f acc = {};
#pragma unroll
    for (int kb = 0; kb < HID; kb += 32) {
      v16bf a = lda_frag(Xh, HID, mstrip, kb, lane);
      v16bf b = ldb_frag(Wb_w1, HID, nt*16, kb, lane);
      acc = wmma_bf16(a, b, acc);
    }
    int n = nt*16 + (lane & 15);
#pragma unroll
    for (int j = 0; j < 8; ++j) {
      int m = mstrip + ((lane >> 4) << 3) + j;
      Tbuf[m*GLR + n] = (bf16_t)fast_tanh(acc[j]);
    }
  }
  __syncthreads();

  // GEMM4: w = T1 @ W_w2^T + b_w2 ; decay = exp(-exp(w))
  for (int nt = 0; nt < 16; ++nt) {
    v8f acc = {};
#pragma unroll
    for (int kb = 0; kb < GLR; kb += 32) {
      v16bf a = lda_frag(Tbuf, GLR, mstrip, kb, lane);
      v16bf b = ldb_frag(Wb_w2, GLR, nt*16, kb, lane);
      acc = wmma_bf16(a, b, acc);
    }
    int n = nt*16 + (lane & 15);
#pragma unroll
    for (int j = 0; j < 8; ++j) {
      int m = mstrip + ((lane >> 4) << 3) + j;
      decayl[m*HID + n] = __expf(-__expf(acc[j] + b_w2[n]));
    }
  }
  __syncthreads();

  // r/g rows at t = 127 only
  {
    const int c = tid;
    float r0 = rwg_bias[c], g0 = rwg_bias[2*HID + c];
    const float* w2r = W_rwg2 + c*3*RLR;
#pragma unroll
    for (int j = 0; j < RLR; ++j) {
      r0 += t0last[j]         * w2r[j];
      g0 += t0last[2*RLR + j] * w2r[2*RLR + j];
    }
    float xv = xb[(LK-1)*HID + c];
    float qv = qr[c];
    xr[c] = xv + (qv - xv) * r0;
    xg[c] = xv + (qv - xv) * g0;
    uu[c] = u[c];
  }
  __syncthreads();
  {
    const int n = tid;
    const float* wr = W_r + n*HID;
    const float* wg = W_g + n*HID;
    float racc = 0.f, gacc = 0.f;
    for (int c2 = 0; c2 < HID; ++c2) {
      racc += xr[c2] * wr[c2];
      gacc += xg[c2] * wg[c2];
    }
    rrow[n] = racc;
    sg[n]   = gacc / (1.f + __expf(-gacc));   // silu(g)
  }
  __syncthreads();

  // ---- RWKV6 scan: thread (h, v) owns S[k][v] for k=0..63 in registers ----
  const int hh = tid >> 6;
  float S[DK];
#pragma unroll
  for (int k = 0; k < DK; ++k) S[k] = 0.f;

  const float* kb_g = kbuf + (size_t)bb * LK * KD;
  const float* vb_g = vbuf + (size_t)bb * LK * VD;

  for (int t = 0; t < LK-1; ++t) {
    // async global -> LDS staging of k_t / v_t rows (ASYNCcnt path)
    async_g2l_b32(ktls + tid, kb_g + t*KD + tid);
    async_g2l_b32(vtls + tid, vb_g + t*VD + tid);
    wait_asynccnt0();
    __syncthreads();
    float vt = vtls[tid];
    const float* dp = decayl + t*HID + hh*DK;
    const float* kp = ktls + hh*DK;
#pragma unroll
    for (int k = 0; k < DK; ++k)
      S[k] = dp[k]*S[k] + kp[k]*vt;
    __syncthreads();
  }
  // final step: o = r . (S + u * (k_t v_t^T))
  {
    const int t = LK-1;
    ktls[tid] = kb_g[t*KD + tid];
    vtls[tid] = vb_g[t*VD + tid];
    __syncthreads();
    float vt = vtls[tid];
    const float* kp = ktls + hh*DK;
    const float* rp = rrow + hh*DK;
    const float* up = uu   + hh*DK;
    float o = 0.f;
#pragma unroll
    for (int k = 0; k < DK; ++k)
      o += rp[k] * (S[k] + up[k]*kp[k]*vt);
    orow[tid] = o;
  }
  __syncthreads();

  // group-norm (per 64-channel head) * silu(g)
  {
    float o = orow[tid];
    const float* op = orow + hh*DV;
    float mu = 0.f;
#pragma unroll 8
    for (int i = 0; i < DV; ++i) mu += op[i];
    mu *= (1.f/DV);
    float var = 0.f;
#pragma unroll 8
    for (int i = 0; i < DV; ++i) { float d = op[i] - mu; var += d*d; }
    var *= (1.f/DV);
    float xn = (o - mu) * rsqrtf(var + EPSV);
    orow2[tid] = (xn * gamma[tid] + beta[tid]) * sg[tid];
  }
  __syncthreads();

  // final projection: out = o @ W_o^T
  {
    const int n = tid;
    const float* wo = W_o + n*HID;
    float acc = 0.f;
    for (int c2 = 0; c2 < VD; ++c2) acc += orow2[c2] * wo[c2];
    out[(size_t)eb*HID + n] = acc;
  }
}

// ===========================================================================
extern "C" void kernel_launch(void* const* d_in, const int* in_sizes, int n_in,
                              void* d_out, int out_size, void* d_ws, size_t ws_size,
                              hipStream_t stream) {
  (void)in_sizes; (void)n_in; (void)out_size; (void)ws_size;
  const float* query    = (const float*)d_in[0];
  const float* keyval   = (const float*)d_in[1];
  const float* mu_rwg   = (const float*)d_in[2];
  const float* W_rwg0   = (const float*)d_in[3];
  const float* W_rwg2   = (const float*)d_in[4];
  const float* rwg_bias = (const float*)d_in[5];
  const float* mu_kv    = (const float*)d_in[6];
  const float* W_kv0    = (const float*)d_in[7];
  const float* W_kv2    = (const float*)d_in[8];
  const float* kv_bias  = (const float*)d_in[9];
  const float* W_r      = (const float*)d_in[10];
  const float* W_w1     = (const float*)d_in[11];
  const float* W_w2     = (const float*)d_in[12];
  const float* b_w2     = (const float*)d_in[13];
  const float* W_k      = (const float*)d_in[14];
  const float* W_v      = (const float*)d_in[15];
  const float* W_g      = (const float*)d_in[16];
  const float* u        = (const float*)d_in[17];
  const float* gamma    = (const float*)d_in[18];
  const float* beta     = (const float*)d_in[19];
  const float* W_o      = (const float*)d_in[20];
  float* out = (float*)d_out;

  float*  kbuf = (float*)d_ws;                         // B*LK*KD f32
  float*  vbuf = kbuf + (size_t)B_*LK*KD;              // B*LK*VD f32
  bf16_t* wbf  = (bf16_t*)(vbuf + (size_t)B_*LK*VD);   // WBF_TOTAL bf16

  const size_t lds1 = (size_t)LK*HID*2*2 + (size_t)LK*GLR*2;               // 147456 B
  const size_t lds2 = (size_t)LK*HID*2 + (size_t)LK*HID*4
                    + (size_t)LK*GLR*2 + (size_t)(128 + 9*256)*4;          // 222720 B

  cvt_weights_kernel<<<(WBF_TOTAL + 255)/256, 256, 0, stream>>>(
      W_kv0, W_kv2, W_k, W_v, W_rwg0, W_rwg2, W_w1, W_w2, wbf);
  kv_proj_kernel<<<B_, 256, lds1, stream>>>(keyval, mu_kv, kv_bias, wbf,
                                            kbuf, vbuf);
  rwkv_fused_kernel<<<EB, 256, lds2, stream>>>(query, keyval, mu_rwg, W_rwg2,
                                               rwg_bias, W_r, b_w2, W_g,
                                               u, gamma, beta, W_o, wbf,
                                               kbuf, vbuf, out);
}
